// TransformerConv_10995116277970
// MI455X (gfx1250) — compile-verified, matrix operand order checked by reading, not run on hardware
//
#include <hip/hip_runtime.h>
#include <hip/hip_bf16.h>

// ---- problem constants (reference: N=50000, E=800000, NIN=256, H=8, C=32) ----
constexpr int kNIN   = 256;   // input features (= K of all GEMMs)
constexpr int kHC    = 256;   // heads * channels (= N of all GEMMs)
constexpr int kH     = 8;     // heads
constexpr int kTM    = 16;    // rows (nodes/edges) per M-tile
constexpr int kLDSP  = 264;   // padded LDS row stride in bf16 elems (avoids bank conflicts)
constexpr int kKST   = 8;     // 256 / 32 k-steps per WMMA chain
constexpr float kInvSqrtC = 0.17677669529663687f; // 1/sqrt(32)

typedef float  v8f   __attribute__((ext_vector_type(8)));
typedef __bf16 v16bf __attribute__((ext_vector_type(16)));

union Frag { uint4 u[2]; v16bf v; };

__device__ __forceinline__ unsigned short f2bf(float f) {
  unsigned b = __float_as_uint(f);
  b += 0x7FFFu + ((b >> 16) & 1u);       // round-to-nearest-even
  return (unsigned short)(b >> 16);
}
// monotonic float<->uint key for atomicMax-based segment max
__device__ __forceinline__ unsigned f2key(float f) {
  unsigned b = __float_as_uint(f);
  return b ^ (unsigned)(((int)b >> 31) | 0x80000000);
}
__device__ __forceinline__ float key2f(unsigned u) {
  unsigned b = (u & 0x80000000u) ? (u ^ 0x80000000u) : ~u;
  return __uint_as_float(b);
}

// A fragment (16x32 bf16, M x K) from a row-major padded LDS tile.
// Per ISA layout: lanes 0-15 hold M=lane, K in [kb, kb+8) and [kb+16, kb+24);
// lanes 16-31 hold M=lane-16, K in [kb+8, kb+16) and [kb+24, kb+32).
__device__ __forceinline__ v16bf loadA(const unsigned short* s, int lane, int kb) {
  int m = lane & 15, half = lane >> 4;
  Frag f;
  f.u[0] = *(const uint4*)&s[m * kLDSP + kb + half * 8];
  f.u[1] = *(const uint4*)&s[m * kLDSP + kb + 16 + half * 8];
  return f.v;
}
// B fragment (32x16 bf16, K x N) from the pre-swizzled pack: 32B contiguous per lane.
__device__ __forceinline__ v16bf loadB(const unsigned short* packW, int mat, int nb,
                                       int ks, int lane) {
  const unsigned short* base =
      packW + ((((size_t)mat * 16 + nb) * 8 + ks) * 32 + lane) * 16;
  Frag f;
  f.u[0] = ((const uint4*)base)[0];
  f.u[1] = ((const uint4*)base)[1];
  return f.v;
}

// ---- kernel 1: convert + swizzle the 5 weight matrices into WMMA B-fragment order
__global__ void pack_weights_kernel(const float* __restrict__ Wq, const float* __restrict__ Wk,
                                    const float* __restrict__ Wv, const float* __restrict__ Ws,
                                    const float* __restrict__ We, unsigned short* __restrict__ packW) {
  int gid = blockIdx.x * blockDim.x + threadIdx.x;
  if (gid >= 5 * 4096) return;              // 5 mats * 16 ntiles * 8 ksteps * 32 lanes
  int lane = gid & 31;
  int ks   = (gid >> 5) & 7;
  int nb   = (gid >> 8) & 15;
  int mat  = gid >> 12;
  const float* W = (mat == 0) ? Wq : (mat == 1) ? Wk : (mat == 2) ? Wv
                 : (mat == 3) ? Ws : We;
  int half = lane >> 4, nn = lane & 15;
  int n = nb * 16 + nn;
  unsigned short* dst =
      packW + ((((size_t)mat * 16 + nb) * 8 + ks) * 32 + lane) * 16;
#pragma unroll
  for (int t = 0; t < 16; ++t) {            // mirror of the A (16-bit) layout, N<->M
    int v = t >> 1, p = t & 1;
    int k = ks * 32 + ((v >= 4) ? 16 : 0) + half * 8 + 2 * (v & 3) + p;
    dst[t] = f2bf(W[(size_t)k * kHC + n]);
  }
}

// ---- kernel 2: q/k/v = x@W+b into workspace, out = x@Ws+bs into d_out
__global__ void __launch_bounds__(512)
node_linear_kernel(const float* __restrict__ x, const unsigned short* __restrict__ packW,
                   const float* __restrict__ bq, const float* __restrict__ bk,
                   const float* __restrict__ bv, const float* __restrict__ bs,
                   float* __restrict__ qf, float* __restrict__ kf,
                   float* __restrict__ vf, float* __restrict__ out, int N) {
  __shared__ unsigned short sX[kTM * kLDSP];
  int tid = threadIdx.x;
  int mbase = blockIdx.x * kTM;
  for (int idx = tid; idx < kTM * kNIN; idx += blockDim.x) {
    int r = idx >> 8, c = idx & 255;
    int row = mbase + r; if (row >= N) row = N - 1;
    sX[r * kLDSP + c] = f2bf(x[(size_t)row * kNIN + c]);
  }
  __syncthreads();
  int lane = tid & 31, nb = tid >> 5;       // 16 waves -> 16 n-tiles
  int half = lane >> 4, nn = lane & 15;
  int n = nb * 16 + nn;
  const float* biases[4] = {bq, bk, bv, bs};
  float*       dsts[4]   = {qf, kf, vf, out};
  for (int mat = 0; mat < 4; ++mat) {
    v8f acc = {};
#pragma unroll
    for (int ks = 0; ks < kKST; ++ks) {
      v16bf a = loadA(sX, lane, ks * 32);
      v16bf b = loadB(packW, mat, nb, ks, lane);
      acc = __builtin_amdgcn_wmma_f32_16x16x32_bf16(false, a, false, b,
                                                    (short)0, acc, false, false);
    }
    float bias = biases[mat][n];
    float* d = dsts[mat];
#pragma unroll
    for (int r = 0; r < 8; ++r) {
      int m = mbase + r + half * 8;
      if (m < N) d[(size_t)m * kHC + n] = acc[r] + bias;
    }
  }
}

// ---- kernel 3: fused edge GEMM (e = edge_attr@We) + alpha = <q_i, k_j+e>/sqrt(C),
//      block reduction makes alpha complete, then atomicMax segment maxima.
__global__ void __launch_bounds__(512)
edge_alpha_kernel(const float* __restrict__ ea, const long long* __restrict__ ei,
                  const unsigned short* __restrict__ packW,
                  const float* __restrict__ qf, const float* __restrict__ kf,
                  float* __restrict__ alpha, unsigned* __restrict__ amaxk,
                  long long E) {
  __shared__ unsigned short sEA[kTM * kLDSP];
  __shared__ float sAl[kTM * kH];
  __shared__ int sSrc[kTM], sDst[kTM];
  int tid = threadIdx.x;
  long long ebase = (long long)blockIdx.x * kTM;
  if (tid < kTM * kH) sAl[tid] = 0.f;
  if (tid < kTM) {
    long long e = ebase + tid; if (e >= E) e = E - 1;
    sSrc[tid] = (int)ei[e];
    sDst[tid] = (int)ei[E + e];
  }
  for (int idx = tid; idx < kTM * kNIN; idx += blockDim.x) {
    int r = idx >> 8, c = idx & 255;
    long long row = ebase + r; if (row >= E) row = E - 1;
    sEA[r * kLDSP + c] = f2bf(ea[row * kNIN + c]);
  }
  __syncthreads();
  int lane = tid & 31, nb = tid >> 5;
  int half = lane >> 4, nn = lane & 15;
  int n = nb * 16 + nn;
  int h = nb >> 1;                          // 2 n-tiles per head (C=32)
  v8f acc = {};
#pragma unroll
  for (int ks = 0; ks < kKST; ++ks) {
    v16bf a = loadA(sEA, lane, ks * 32);
    v16bf b = loadB(packW, 4, nb, ks, lane);
    acc = __builtin_amdgcn_wmma_f32_16x16x32_bf16(false, a, false, b,
                                                  (short)0, acc, false, false);
  }
#pragma unroll
  for (int r = 0; r < 8; ++r) {
    int m = r + half * 8;
    float kj = kf[(size_t)sSrc[m] * kHC + n] + acc[r];   // L2-resident gather
    float p  = qf[(size_t)sDst[m] * kHC + n] * kj;
#pragma unroll
    for (int off = 8; off > 0; off >>= 1) p += __shfl_xor(p, off, 32);
    if (nn == 0) atomicAdd(&sAl[m * kH + h], p);         // combine the 2 half-head tiles
  }
  __syncthreads();
  if (tid < kTM * kH) {
    int m = tid >> 3, hh = tid & 7;
    long long e = ebase + m;
    if (e < E) {
      float a = sAl[tid] * kInvSqrtC;
      alpha[e * kH + hh] = a;                            // complete value
      atomicMax(&amaxk[(size_t)sDst[m] * kH + hh], f2key(a));
    }
  }
}

// ---- kernel 4: ex = exp(alpha - amax[dst]); denom[dst] += ex
__global__ void softmax_norm_kernel(float* __restrict__ alpha, const long long* __restrict__ ei,
                                    const unsigned* __restrict__ amaxk,
                                    float* __restrict__ denom, long long EH, long long E) {
  long long idx = (long long)blockIdx.x * blockDim.x + threadIdx.x;
  if (idx >= EH) return;
  long long e = idx >> 3; int h = (int)(idx & 7);
  int d = (int)ei[E + e];
  float amax = key2f(amaxk[(size_t)d * kH + h]);
  float ex = __expf(alpha[idx] - amax);
  alpha[idx] = ex;
  atomicAdd(&denom[(size_t)d * kH + h], ex);
}

// ---- kernel 5: recompute edge GEMM, scatter coeff*(v[src]+e) into out[dst]
__global__ void __launch_bounds__(512)
edge_scatter_kernel(const float* __restrict__ ea, const long long* __restrict__ ei,
                    const unsigned short* __restrict__ packW,
                    const float* __restrict__ vf, const float* __restrict__ alpha,
                    const float* __restrict__ denom, float* __restrict__ out,
                    long long E) {
  __shared__ unsigned short sEA[kTM * kLDSP];
  __shared__ int sSrc[kTM], sDst[kTM];
  int tid = threadIdx.x;
  long long ebase = (long long)blockIdx.x * kTM;
  if (tid < kTM) {
    long long e = ebase + tid; if (e >= E) e = E - 1;
    sSrc[tid] = (int)ei[e];
    sDst[tid] = (int)ei[E + e];
  }
  for (int idx = tid; idx < kTM * kNIN; idx += blockDim.x) {
    int r = idx >> 8, c = idx & 255;
    long long row = ebase + r; if (row >= E) row = E - 1;
    sEA[r * kLDSP + c] = f2bf(ea[row * kNIN + c]);
  }
  __syncthreads();
  int lane = tid & 31, nb = tid >> 5;
  int half = lane >> 4, nn = lane & 15;
  int n = nb * 16 + nn;
  int h = nb >> 1;
  v8f acc = {};
#pragma unroll
  for (int ks = 0; ks < kKST; ++ks) {
    v16bf a = loadA(sEA, lane, ks * 32);
    v16bf b = loadB(packW, 4, nb, ks, lane);
    acc = __builtin_amdgcn_wmma_f32_16x16x32_bf16(false, a, false, b,
                                                  (short)0, acc, false, false);
  }
#pragma unroll
  for (int r = 0; r < 8; ++r) {
    int m = r + half * 8;
    long long e = ebase + m;
    int src = sSrc[m], dst = sDst[m];
    float ex  = alpha[e * kH + h];
    float den = denom[(size_t)dst * kH + h];
    float coeff = ex / den;
    float vj = vf[(size_t)src * kHC + n] + acc[r];
    if (e < E) atomicAdd(&out[(size_t)dst * kHC + n], coeff * vj);
  }
}

extern "C" void kernel_launch(void* const* d_in, const int* in_sizes, int n_in,
                              void* d_out, int out_size, void* d_ws, size_t ws_size,
                              hipStream_t stream) {
  const float*     x  = (const float*)d_in[0];
  const long long* ei = (const long long*)d_in[1];   // [2, E] int64
  const float*     ea = (const float*)d_in[2];
  const float* Wq = (const float*)d_in[3];
  const float* bq = (const float*)d_in[4];
  const float* Wk = (const float*)d_in[5];
  const float* bk = (const float*)d_in[6];
  const float* Wv = (const float*)d_in[7];
  const float* bv = (const float*)d_in[8];
  const float* We = (const float*)d_in[9];
  const float* Ws = (const float*)d_in[10];
  const float* bs = (const float*)d_in[11];
  float* out = (float*)d_out;

  const int       N = in_sizes[0] / kNIN;
  const long long E = (long long)(in_sizes[2] / kNIN);

  // ---- workspace carve-out (aligned 256B): qf,kf,vf (f32), packed bf16 weights,
  //      alpha [E,H], then a contiguous zero region {amaxk, denom} ----
  char* ws = (char*)d_ws;
  size_t off = 0;
  auto carve = [&](size_t bytes) -> void* {
    void* p = ws + off;
    off = (off + bytes + 255) & ~(size_t)255;
    return p;
  };
  float*          qf    = (float*)carve((size_t)N * kHC * sizeof(float));
  float*          kf    = (float*)carve((size_t)N * kHC * sizeof(float));
  float*          vf    = (float*)carve((size_t)N * kHC * sizeof(float));
  unsigned short* packW = (unsigned short*)carve((size_t)5 * 256 * 256 * sizeof(unsigned short));
  float*          alpha = (float*)carve((size_t)E * kH * sizeof(float));
  char*           zbase = ws + off;
  unsigned*       amaxk = (unsigned*)carve((size_t)N * kH * sizeof(unsigned));
  float*          denom = (float*)carve((size_t)N * kH * sizeof(float));
  size_t zbytes = (size_t)((ws + off) - zbase);
  hipMemsetAsync(zbase, 0, zbytes, stream);   // amaxk=key(-NaN) floor, denom=0

  pack_weights_kernel<<<80, 256, 0, stream>>>(Wq, Wk, Wv, Ws, We, packW);

  int nblocks = (N + kTM - 1) / kTM;
  node_linear_kernel<<<nblocks, 512, 0, stream>>>(x, packW, bq, bk, bv, bs,
                                                  qf, kf, vf, out, N);

  int eblocks = (int)((E + kTM - 1) / kTM);
  edge_alpha_kernel<<<eblocks, 512, 0, stream>>>(ea, ei, packW, qf, kf,
                                                 alpha, amaxk, E);

  long long EH = E * kH;
  softmax_norm_kernel<<<(int)((EH + 255) / 256), 256, 0, stream>>>(alpha, ei, amaxk,
                                                                   denom, EH, E);

  edge_scatter_kernel<<<eblocks, 512, 0, stream>>>(ea, ei, packW, vf, alpha,
                                                   denom, out, E);
}
